// LSTM_autoreg_torchscript_35442070126824
// MI455X (gfx1250) — compile-verified, matrix operand order checked by reading, not run on hardware
//
#include <hip/hip_runtime.h>
#include <hip/hip_bf16.h>
#include <cstdint>

// ---------------- problem constants ----------------
constexpr int B_    = 8192;
constexpr int T_    = 60;
constexpr int NX_   = 4;
constexpr int NMEM_ = 16;
constexpr int NH_   = 64;    // hidden (both LSTMs)
constexpr int G_    = 256;   // 4*NH gate dim
constexpr int K1_   = 160;   // LSTM1: 96 (x padded from 80) + 64 (h)
constexpr int K2_   = 128;   // LSTM2: 64 (r1) + 64 (h)
constexpr int XW_   = 96;    // padded x width for LSTM1
constexpr int ROWS_ = 64;    // batch rows per workgroup
constexpr int NTH_  = 256;   // 8 waves (wave32): 4 row-blocks x 2 N-halves

typedef __attribute__((ext_vector_type(16))) _Float16 v16h;
typedef __attribute__((ext_vector_type(8)))  float    v8f;

// ---------------- workspace layout (bytes) ----------------
constexpr size_t OFF_H01  = 0;                                          // f16 [B][64]
constexpr size_t OFF_C01  = OFF_H01 + (size_t)B_ * NH_ * 2;             // f32 [B][64]
constexpr size_t OFF_H02  = OFF_C01 + (size_t)B_ * NH_ * 4;             // f16 [B][64]
constexpr size_t OFF_C02  = OFF_H02 + (size_t)B_ * NH_ * 2;             // f32 [B][64]
constexpr size_t OFF_R1   = OFF_C02 + (size_t)B_ * NH_ * 4;             // f16 [T][B][64]
constexpr size_t OFF_HL   = OFF_R1  + (size_t)T_ * B_ * NH_ * 2;        // f32 [B][64]
constexpr size_t OFF_XCAT = OFF_HL  + (size_t)B_ * NH_ * 4;             // f16 [T][B][96]

// ---------------- helpers ----------------
__device__ __forceinline__ float sigmoidf_(float x) {
    return 1.0f / (1.0f + __expf(-x));
}

// 16-half WMMA fragment slice: element e -> k = kbase + 16*(e/8) + (e%8) + 8*(lane/16).
// Caller passes q = base + row*stride + kbase + 8*(lane>>4); loads q[0..7] and q[16..23].
__device__ __forceinline__ v16h ld_frag(const _Float16* q) {
    union { v16h v; uint4 u[2]; } r;
    r.u[0] = *reinterpret_cast<const uint4*>(q);
    r.u[1] = *reinterpret_cast<const uint4*>(q + 16);
    return r.v;
}

__device__ __forceinline__ v8f wmma_f16(v16h a, v16h b, v8f c) {
    return __builtin_amdgcn_wmma_f32_16x16x32_f16(false, a, false, b, (short)0, c, false, false);
}

// Gate GEMM for one wave: 8 accumulator tiles (gates i,f,g,o x 2 j-tiles), KT k-tiles.
// aRow = &sXH[arow*KSTR + 8*lh]; wCol = &sW[(jt0*16 + ln)*KSTR + 8*lh].
// acc[a] (a = gate*2+u) accumulates ntile = gate*4 + jt0 + u; B offset relative to wCol
// is ((gate*4) + u)*16*KSTR (jt0 folded into wCol). Software-pipelined: next fragment
// loads issue before each WMMA so ds waits overlap matrix execution.
template<int KT, int KSTR>
__device__ __forceinline__ void gemm_gates(v8f acc[8], const _Float16* aRow, const _Float16* wCol) {
    v16h afr = ld_frag(aRow);
    v16h bfr = ld_frag(wCol);
    v16h na = afr, nb = bfr;
#pragma unroll
    for (int it = 0; it < KT * 8; ++it) {
        const int kt = it >> 3, a = it & 7;
        if (a < 7) {
            const int an = a + 1;
            nb = ld_frag(wCol + (((an >> 1) * 4 + (an & 1)) * 16) * KSTR + kt * 32);
        } else if (kt + 1 < KT) {
            nb = ld_frag(wCol + (kt + 1) * 32);
            na = ld_frag(aRow + (kt + 1) * 32);
        }
        acc[a] = wmma_f16(afr, bfr, acc[a]);
        bfr = nb;
        if (a == 7) afr = na;
    }
}

// ---------------- kernel 0: initial states ----------------
__global__ __launch_bounds__(256) void prep_kernel(
    const float* __restrict__ aux,
    const float* __restrict__ Ws1, const float* __restrict__ bs1,
    const float* __restrict__ Ws2, const float* __restrict__ bs2,
    const float* __restrict__ Wt1, const float* __restrict__ bt1,
    const float* __restrict__ Wt2, const float* __restrict__ bt2,
    _Float16* __restrict__ h01, float* __restrict__ c01,
    _Float16* __restrict__ h02, float* __restrict__ c02)
{
    int idx = blockIdx.x * blockDim.x + threadIdx.x;
    if (idx >= B_ * NH_) return;
    int b = idx >> 6, j = idx & 63;
    const float* a = aux + (size_t)b * 3;
    float h1 = tanhf(a[0]*Ws1[j*3+0] + a[1]*Ws1[j*3+1] + a[2]*Ws1[j*3+2] + bs1[j]);
    float c1 = tanhf(a[0]*Ws2[j*3+0] + a[1]*Ws2[j*3+1] + a[2]*Ws2[j*3+2] + bs2[j]);
    float toa = a[1];
    float h2 = toa * Wt1[j] + bt1[j];   // linear (no tanh) per reference
    float c2 = toa * Wt2[j] + bt2[j];
    h01[idx] = (_Float16)h1;  c01[idx] = c1;
    h02[idx] = (_Float16)h2;  c02[idx] = c2;
}

// ---------------- kernel 1: xcat = [tanh(x@Wi^T+bi) | mem | 0pad] (f16, t-major) ----------------
__global__ __launch_bounds__(256) void xcat_kernel(
    const float* __restrict__ inputs_main,  // [B][T][4]
    const float* __restrict__ rnn1_mem,     // [B][T][16]
    const float* __restrict__ Wi,           // [64][4]
    const float* __restrict__ bi,           // [64]
    _Float16* __restrict__ xcat)            // [T][B][96]
{
    size_t idx = (size_t)blockIdx.x * blockDim.x + threadIdx.x;
    if (idx >= (size_t)B_ * T_ * XW_) return;
    int c = (int)(idx % XW_);
    size_t tb = idx / XW_;
    int b = (int)(tb % B_);
    int t = (int)(tb / B_);
    float v;
    if (c < NH_) {
        const float* im = inputs_main + ((size_t)b * T_ + t) * NX_;
        v = tanhf(im[0]*Wi[c*4+0] + im[1]*Wi[c*4+1] + im[2]*Wi[c*4+2] + im[3]*Wi[c*4+3] + bi[c]);
    } else if (c < 80) {
        v = rnn1_mem[((size_t)b * T_ + t) * NMEM_ + (c - NH_)];
    } else {
        v = 0.0f;
    }
    xcat[idx] = (_Float16)v;
}

// ---------------- kernel 2: LSTM1 (reverse time) ----------------
__global__ __launch_bounds__(NTH_) void lstm1_kernel(
    const _Float16* __restrict__ xcat,      // [T][B][96] f16
    const float* __restrict__ Wih1,         // [256][80]
    const float* __restrict__ Whh1,         // [256][64]
    const float* __restrict__ bih1,
    const float* __restrict__ bhh1,
    const _Float16* __restrict__ h0,        // [B][64] f16
    const float* __restrict__ c0,           // [B][64] f32
    _Float16* __restrict__ r1out)           // [T][B][64] f16
{
    __shared__ __align__(16) _Float16 sW[G_ * K1_];     // 80 KB combined [Wih|0pad|Whh]
    __shared__ __align__(16) _Float16 sXH[ROWS_ * K1_]; // 20 KB per-step [x(96) | h(64)]

    const int tid  = threadIdx.x;
    const int lane = tid & 31;
    const int wave = tid >> 5;
    const int lh   = lane >> 4;
    const int ln   = lane & 15;
    const int rb   = (wave >> 1) * 16;        // row block within tile
    const int jt0  = (wave & 1) * 2;          // j-tile pair handled by this wave
    const int rowbase = blockIdx.x * ROWS_;

    for (int idx = tid; idx < G_ * K1_; idx += NTH_) {
        int n = idx / K1_, k = idx - n * K1_;
        float v = (k < 80) ? Wih1[n*80 + k] : ((k < XW_) ? 0.0f : Whh1[n*64 + (k - XW_)]);
        sW[idx] = (_Float16)v;
    }
    for (int idx = tid; idx < ROWS_ * NH_; idx += NTH_) {
        int r = idx >> 6, j = idx & 63;
        sXH[r * K1_ + XW_ + j] = h0[(size_t)(rowbase + r) * NH_ + j];
    }

    // c state in accumulator layout: cstate[u][r] = c[row=rb+r+8*lh][j=(jt0+u)*16+ln]
    v8f cstate[2];
#pragma unroll
    for (int u = 0; u < 2; ++u)
#pragma unroll
        for (int r = 0; r < 8; ++r) {
            int row = rowbase + rb + r + 8 * lh;
            cstate[u][r] = c0[(size_t)row * NH_ + (jt0 + u) * 16 + ln];
        }

    const _Float16* aRow = &sXH[(rb + ln) * K1_ + 8 * lh];
    const _Float16* wCol = &sW[(jt0 * 16 + ln) * K1_ + 8 * lh];
    float bi_[2], bf_[2], bg_[2], bo_[2];
#pragma unroll
    for (int u = 0; u < 2; ++u) {
        bi_[u] = bih1[  0 + (jt0+u)*16 + ln] + bhh1[  0 + (jt0+u)*16 + ln];
        bf_[u] = bih1[ 64 + (jt0+u)*16 + ln] + bhh1[ 64 + (jt0+u)*16 + ln];
        bg_[u] = bih1[128 + (jt0+u)*16 + ln] + bhh1[128 + (jt0+u)*16 + ln];
        bo_[u] = bih1[192 + (jt0+u)*16 + ln] + bhh1[192 + (jt0+u)*16 + ln];
    }

    // x-tile staging addresses (4 threads per row, 3x b128 each) + first prefetch
    const int srow = tid >> 2, soff = (tid & 3) * 24;
    uint4* d4 = reinterpret_cast<uint4*>(&sXH[srow * K1_ + soff]);
    uint4 px0, px1, px2;
    {
        const uint4* s4 = reinterpret_cast<const uint4*>(
            xcat + ((size_t)(T_ - 1) * B_ + rowbase + srow) * XW_ + soff);
        px0 = s4[0]; px1 = s4[1]; px2 = s4[2];
    }

    for (int t = T_ - 1; t >= 0; --t) {
        // ---- commit prefetched x tile, then immediately issue next prefetch ----
        d4[0] = px0; d4[1] = px1; d4[2] = px2;
        if (t > 0) {
            const uint4* s4 = reinterpret_cast<const uint4*>(
                xcat + ((size_t)(t - 1) * B_ + rowbase + srow) * XW_ + soff);
            px0 = s4[0]; px1 = s4[1]; px2 = s4[2];   // latency hidden behind GEMM+epilogue
        }
        __syncthreads();   // x stores + previous h writes visible to all waves

        // ---- gates: [16 rows] x [8 gate tiles] over K1 ----
        v8f acc[8];
#pragma unroll
        for (int a = 0; a < 8; ++a)
#pragma unroll
            for (int e = 0; e < 8; ++e) acc[a][e] = 0.0f;
        gemm_gates<5, K1_>(acc, aRow, wCol);
        __syncthreads();   // all x/h reads for step t done

        // ---- epilogue: activations, c/h update, h -> LDS + r1 -> global ----
#pragma unroll
        for (int u = 0; u < 2; ++u)
#pragma unroll
            for (int r = 0; r < 8; ++r) {
                float ig = sigmoidf_(acc[0 + u][r] + bi_[u]);
                float fg = sigmoidf_(acc[2 + u][r] + bf_[u]);
                float gg = tanhf    (acc[4 + u][r] + bg_[u]);
                float og = sigmoidf_(acc[6 + u][r] + bo_[u]);
                float c  = fg * cstate[u][r] + ig * gg;
                cstate[u][r] = c;
                float h = og * tanhf(c);
                int rloc = rb + r + 8 * lh;
                _Float16 hh = (_Float16)h;
                sXH[rloc * K1_ + XW_ + (jt0 + u) * 16 + ln] = hh;
                r1out[((size_t)t * B_ + rowbase + rloc) * NH_ + (jt0 + u) * 16 + ln] = hh;
            }
        // next iteration's pre-GEMM barrier orders h writes + x stores before reads
    }
}

// ---------------- kernel 3: LSTM2 (forward time) + Wl projection ----------------
__global__ __launch_bounds__(NTH_) void lstm2_kernel(
    const _Float16* __restrict__ r1in,      // [T][B][64] f16
    const float* __restrict__ Wih2,         // [256][64]
    const float* __restrict__ Whh2,         // [256][64]
    const float* __restrict__ bih2,
    const float* __restrict__ bhh2,
    const float* __restrict__ Wl,           // [16][64]
    const float* __restrict__ bl,           // [16]
    const _Float16* __restrict__ h0,        // [B][64] f16
    const float* __restrict__ c0,           // [B][64] f32
    float* __restrict__ new_mem,            // [B][T][16] f32 (inside d_out)
    float* __restrict__ hL)                 // [B][64] f32
{
    __shared__ __align__(16) _Float16 sW[G_ * K2_];     // 64 KB combined [Wih2|Whh2]
    __shared__ __align__(16) _Float16 sXH[ROWS_ * K2_]; // 16 KB [x(64) | h(64)]
    __shared__ __align__(16) _Float16 sWl[16 * NH_];    // 2 KB

    const int tid  = threadIdx.x;
    const int lane = tid & 31;
    const int wave = tid >> 5;
    const int lh   = lane >> 4;
    const int ln   = lane & 15;
    const int rb   = (wave >> 1) * 16;
    const int jt0  = (wave & 1) * 2;
    const int rowbase = blockIdx.x * ROWS_;

    for (int idx = tid; idx < G_ * K2_; idx += NTH_) {
        int n = idx / K2_, k = idx - n * K2_;
        float v = (k < 64) ? Wih2[n*64 + k] : Whh2[n*64 + (k - 64)];
        sW[idx] = (_Float16)v;
    }
    for (int idx = tid; idx < 16 * NH_; idx += NTH_) sWl[idx] = (_Float16)Wl[idx];
    for (int idx = tid; idx < ROWS_ * NH_; idx += NTH_) {
        int r = idx >> 6, j = idx & 63;
        sXH[r * K2_ + 64 + j] = h0[(size_t)(rowbase + r) * NH_ + j];
    }

    v8f cstate[2];
#pragma unroll
    for (int u = 0; u < 2; ++u)
#pragma unroll
        for (int r = 0; r < 8; ++r) {
            int row = rowbase + rb + r + 8 * lh;
            cstate[u][r] = c0[(size_t)row * NH_ + (jt0 + u) * 16 + ln];
        }

    const _Float16* aRow = &sXH[(rb + ln) * K2_ + 8 * lh];
    const _Float16* wCol = &sW[(jt0 * 16 + ln) * K2_ + 8 * lh];
    float bi_[2], bf_[2], bg_[2], bo_[2];
#pragma unroll
    for (int u = 0; u < 2; ++u) {
        bi_[u] = bih2[  0 + (jt0+u)*16 + ln] + bhh2[  0 + (jt0+u)*16 + ln];
        bf_[u] = bih2[ 64 + (jt0+u)*16 + ln] + bhh2[ 64 + (jt0+u)*16 + ln];
        bg_[u] = bih2[128 + (jt0+u)*16 + ln] + bhh2[128 + (jt0+u)*16 + ln];
        bo_[u] = bih2[192 + (jt0+u)*16 + ln] + bhh2[192 + (jt0+u)*16 + ln];
    }
    const float blv = bl[ln];

    // x-tile staging addresses (4 threads per row, 2x b128 each) + first prefetch
    const int srow = tid >> 2, soff = (tid & 3) * 16;
    uint4* d4 = reinterpret_cast<uint4*>(&sXH[srow * K2_ + soff]);
    uint4 px0, px1;
    {
        const uint4* s4 = reinterpret_cast<const uint4*>(
            r1in + ((size_t)0 * B_ + rowbase + srow) * NH_ + soff);
        px0 = s4[0]; px1 = s4[1];
    }

    for (int t = 0; t < T_; ++t) {
        // ---- commit prefetched x tile, then immediately issue next prefetch ----
        d4[0] = px0; d4[1] = px1;
        if (t + 1 < T_) {
            const uint4* s4 = reinterpret_cast<const uint4*>(
                r1in + ((size_t)(t + 1) * B_ + rowbase + srow) * NH_ + soff);
            px0 = s4[0]; px1 = s4[1];   // latency hidden behind GEMM+epilogue
        }
        __syncthreads();   // x stores + previous h writes visible to all waves

        // ---- gates ----
        v8f acc[8];
#pragma unroll
        for (int a = 0; a < 8; ++a)
#pragma unroll
            for (int e = 0; e < 8; ++e) acc[a][e] = 0.0f;
        gemm_gates<4, K2_>(acc, aRow, wCol);
        __syncthreads();   // all x/h reads for step t done

        // ---- epilogue: h/c update ----
#pragma unroll
        for (int u = 0; u < 2; ++u)
#pragma unroll
            for (int r = 0; r < 8; ++r) {
                float ig = sigmoidf_(acc[0 + u][r] + bi_[u]);
                float fg = sigmoidf_(acc[2 + u][r] + bf_[u]);
                float gg = tanhf    (acc[4 + u][r] + bg_[u]);
                float og = sigmoidf_(acc[6 + u][r] + bo_[u]);
                float c  = fg * cstate[u][r] + ig * gg;
                cstate[u][r] = c;
                float h = og * tanhf(c);
                int rloc = rb + r + 8 * lh;
                sXH[rloc * K2_ + 64 + (jt0 + u) * 16 + ln] = (_Float16)h;
                if (t == T_ - 1)
                    hL[(size_t)(rowbase + rloc) * NH_ + (jt0 + u) * 16 + ln] = h;
            }
        __syncthreads();   // full h row visible for lat projection

        // ---- lat = h @ Wl^T + bl (even waves, one 16x16 tile per row block) ----
        if ((wave & 1) == 0) {
            v8f lacc;
#pragma unroll
            for (int e = 0; e < 8; ++e) lacc[e] = 0.0f;
#pragma unroll
            for (int kt = 0; kt < 2; ++kt) {
                v16h afrag = ld_frag(&sXH[(rb + ln) * K2_ + 64 + kt * 32 + 8 * lh]);
                v16h bfrag = ld_frag(&sWl[ln * NH_ + kt * 32 + 8 * lh]);
                lacc = wmma_f16(afrag, bfrag, lacc);
            }
#pragma unroll
            for (int r = 0; r < 8; ++r) {
                int rloc = rb + r + 8 * lh;
                new_mem[((size_t)(rowbase + rloc) * T_ + t) * NMEM_ + ln] = lacc[r] + blv;
            }
        }
        // next iteration's pre-GEMM barrier orders lat reads before h overwrite
        // (x staging touches x-section only; lat reads h-section -> disjoint)
    }
}

// ---------------- kernel 4: out = new_mem @ Wo^T + bo ----------------
__global__ __launch_bounds__(256) void out_kernel(
    const float* __restrict__ new_mem, const float* __restrict__ Wo,
    const float* __restrict__ bo, float* __restrict__ out)
{
    int idx = blockIdx.x * blockDim.x + threadIdx.x;   // (b,t) flat
    if (idx >= B_ * T_) return;
    const float* lat = new_mem + (size_t)idx * NMEM_;
    float l[16];
#pragma unroll
    for (int k = 0; k < 16; ++k) l[k] = lat[k];
#pragma unroll
    for (int j = 0; j < 4; ++j) {
        float s = bo[j];
#pragma unroll
        for (int k = 0; k < 16; ++k) s += l[k] * Wo[j * 16 + k];
        out[(size_t)idx * 4 + j] = s;
    }
}

// ---------------- kernel 5: out_sfc = hL @ Wso^T + bso ----------------
__global__ __launch_bounds__(256) void sfc_kernel(
    const float* __restrict__ hL, const float* __restrict__ Wso,
    const float* __restrict__ bso, float* __restrict__ out_sfc)
{
    int b = blockIdx.x * blockDim.x + threadIdx.x;
    if (b >= B_) return;
    const float* h = hL + (size_t)b * NH_;
    for (int j = 0; j < 3; ++j) {
        float s = bso[j];
        for (int k = 0; k < NH_; ++k) s += h[k] * Wso[j * NH_ + k];
        out_sfc[(size_t)b * 3 + j] = s;
    }
}

// ---------------- launcher ----------------
extern "C" void kernel_launch(void* const* d_in, const int* in_sizes, int n_in,
                              void* d_out, int out_size, void* d_ws, size_t ws_size,
                              hipStream_t stream) {
    const float* inputs_main = (const float*)d_in[0];
    const float* inputs_aux  = (const float*)d_in[1];
    const float* rnn1_mem    = (const float*)d_in[2];
    const float* Wi   = (const float*)d_in[3];
    const float* bi   = (const float*)d_in[4];
    const float* Ws1  = (const float*)d_in[5];
    const float* bs1  = (const float*)d_in[6];
    const float* Ws2  = (const float*)d_in[7];
    const float* bs2  = (const float*)d_in[8];
    const float* Wt1  = (const float*)d_in[9];
    const float* bt1  = (const float*)d_in[10];
    const float* Wt2  = (const float*)d_in[11];
    const float* bt2  = (const float*)d_in[12];
    const float* Wih1 = (const float*)d_in[13];
    const float* Whh1 = (const float*)d_in[14];
    const float* bih1 = (const float*)d_in[15];
    const float* bhh1 = (const float*)d_in[16];
    const float* Wih2 = (const float*)d_in[17];
    const float* Whh2 = (const float*)d_in[18];
    const float* bih2 = (const float*)d_in[19];
    const float* bhh2 = (const float*)d_in[20];
    const float* Wl   = (const float*)d_in[21];
    const float* bl   = (const float*)d_in[22];
    const float* Wo   = (const float*)d_in[23];
    const float* bo   = (const float*)d_in[24];
    const float* Wso  = (const float*)d_in[25];
    const float* bso  = (const float*)d_in[26];

    char* ws = (char*)d_ws;
    _Float16* h01  = (_Float16*)(ws + OFF_H01);
    float*    c01  = (float*)   (ws + OFF_C01);
    _Float16* h02  = (_Float16*)(ws + OFF_H02);
    float*    c02  = (float*)   (ws + OFF_C02);
    _Float16* r1   = (_Float16*)(ws + OFF_R1);
    float*    hL   = (float*)   (ws + OFF_HL);
    _Float16* xcat = (_Float16*)(ws + OFF_XCAT);

    float* out     = (float*)d_out;                      // [B][T][4]
    float* out_sfc = out + (size_t)B_ * T_ * 4;          // [B][3]
    float* new_mem = out_sfc + (size_t)B_ * 3;           // [B][T][16]

    prep_kernel<<<(B_ * NH_ + 255) / 256, 256, 0, stream>>>(
        inputs_aux, Ws1, bs1, Ws2, bs2, Wt1, bt1, Wt2, bt2, h01, c01, h02, c02);

    xcat_kernel<<<(int)(((size_t)B_ * T_ * XW_ + 255) / 256), 256, 0, stream>>>(
        inputs_main, rnn1_mem, Wi, bi, xcat);

    lstm1_kernel<<<B_ / ROWS_, NTH_, 0, stream>>>(
        xcat, Wih1, Whh1, bih1, bhh1, h01, c01, r1);

    lstm2_kernel<<<B_ / ROWS_, NTH_, 0, stream>>>(
        r1, Wih2, Whh2, bih2, bhh2, Wl, bl, h02, c02, new_mem, hL);

    out_kernel<<<(B_ * T_ + 255) / 256, 256, 0, stream>>>(new_mem, Wo, bo, out);
    sfc_kernel<<<(B_ + 255) / 256, 256, 0, stream>>>(hL, Wso, bso, out_sfc);
}
